// RW_layer_60662118089061
// MI455X (gfx1250) — matrix operand order repaired and checked by reading, not compiled, not run on hardware
//
#include <hip/hip_runtime.h>
#include <hip/hip_bf16.h>
#include <math.h>

typedef __attribute__((ext_vector_type(16))) _Float16 v16h;
typedef __attribute__((ext_vector_type(8)))  _Float16 v8h;
typedef __attribute__((ext_vector_type(8)))  float    v8f;

__device__ __forceinline__ v8f v8f_zero() {
  v8f z;
#pragma unroll
  for (int i = 0; i < 8; ++i) z[i] = 0.0f;
  return z;
}

// ---------------------------------------------------------------------------
// Kernel 1: hidden-prototype side (tiny). Builds symmetric A from triu-packed
// relu(adj_hidden), propagates z for 3 steps, writes zs_step[h,o] summed over m
// as f16 in layout zsT[step][o][h] (B-fragment friendly). Zeroes BN accums.
// ---------------------------------------------------------------------------
__global__ __launch_bounds__(256) void k_hidden(
    const float* __restrict__ fh,    // features_hidden (16,64,32)
    const float* __restrict__ ah,    // adj_hidden (120,32)
    _Float16* __restrict__ zsT,      // [3][32][64]
    float* __restrict__ accum)       // 64 floats (sum[32], sumsq[32])
{
  __shared__ float A[16][16][32];
  const int tid = threadIdx.x;
  for (int i = tid; i < 16 * 16 * 32; i += 256) ((float*)A)[i] = 0.0f;
  __syncthreads();
  for (int e = tid; e < 120 * 32; e += 256) {
    int p = e >> 5, o = e & 31;
    // row-major triu(k=1) enumeration: row i has (15-i) pairs
    int i = 0, off = 0;
    while (off + (15 - i) <= p) { off += 15 - i; ++i; }
    int j = i + 1 + (p - off);
    float a = ah[p * 32 + o];
    a = a > 0.0f ? a : 0.0f;
    A[i][j][o] = a;
    A[j][i][o] = a;
  }
  if (tid < 64) accum[tid] = 0.0f;
  __syncthreads();

  // 2048 (h,o) pairs, 8 per thread
#pragma unroll
  for (int q = 0; q < 8; ++q) {
    int pi = tid + q * 256;
    int o = pi >> 6, hh = pi & 63;
    float z[16];
#pragma unroll
    for (int m = 0; m < 16; ++m) z[m] = fh[(m * 64 + hh) * 32 + o];
#pragma unroll
    for (int step = 0; step < 3; ++step) {
      float s = 0.0f;
#pragma unroll
      for (int m = 0; m < 16; ++m) s += z[m];
      zsT[(step * 32 + o) * 64 + hh] = (_Float16)s;
      if (step < 2) {
        float zn[16];
#pragma unroll
        for (int m = 0; m < 16; ++m) {
          float t = 0.0f;
#pragma unroll
          for (int p2 = 0; p2 < 16; ++p2) t += A[m][p2][o] * z[p2];
          zn[m] = t;
        }
#pragma unroll
        for (int m = 0; m < 16; ++m) z[m] = zn[m];
      }
    }
  }
}

// ---------------------------------------------------------------------------
// Kernel 2: main. One wave per batch, 16 waves (16 batches) per workgroup.
// ---------------------------------------------------------------------------
__global__ __launch_bounds__(512) void k_rw_main(
    const float* __restrict__ adj,    // (B,16,16)
    const float* __restrict__ feat,   // (B,16,128)
    const float* __restrict__ Win,    // (128,64)
    const float* __restrict__ bin,    // (64)
    const _Float16* __restrict__ zsT, // [3][32][64]
    float* __restrict__ accum,        // 64
    float* __restrict__ out_raw)      // (B,32), pre-BN
{
  __shared__ __align__(32) _Float16 WT[64][128];      // WT[n][k] = W_in[k][n]
  __shared__ __align__(32) _Float16 XT[16][64][16];   // [wave][col h][row s]
  __shared__ __align__(32) float    XS[16][3][64];    // [wave][step][col h]

  const int tid  = threadIdx.x;
  const int wave = tid >> 5;
  const int lane = tid & 31;
  const int h    = lane >> 4;   // lane-half
  const int n    = lane & 15;   // N / M index within half
  const long b   = (long)blockIdx.x * 16 + wave;

  // stage W transposed as f16
  for (int i = tid; i < 64 * 128; i += 512) {
    int k = i & 127, c = i >> 7;
    WT[c][k] = (_Float16)Win[k * 64 + c];
  }
  __syncthreads();

  // adj A-fragment (16x16 padded to 16x32: elems 8..15 are K=16..31 -> zero)
  v16h adjA;
  {
    const float4* ar = (const float4*)(adj + b * 256 + n * 16 + h * 8);
    float4 a0 = ar[0], a1 = ar[1];
    adjA[0] = (_Float16)a0.x; adjA[1] = (_Float16)a0.y;
    adjA[2] = (_Float16)a0.z; adjA[3] = (_Float16)a0.w;
    adjA[4] = (_Float16)a1.x; adjA[5] = (_Float16)a1.y;
    adjA[6] = (_Float16)a1.z; adjA[7] = (_Float16)a1.w;
#pragma unroll
    for (int e = 8; e < 16; ++e) adjA[e] = (_Float16)0.0f;
  }

  // X0 = sigmoid(F @ W + b): four 16x16 C tiles (nt), K=128 -> 4 kt
  v8f Xc[4];
#pragma unroll
  for (int nt = 0; nt < 4; ++nt) Xc[nt] = v8f_zero();
#pragma unroll
  for (int kt = 0; kt < 4; ++kt) {
    v16h Fa;
    const float* Fr = feat + (b * 16 + n) * 128 + kt * 32 + h * 8;
    float4 f0 = *(const float4*)(Fr);
    float4 f1 = *(const float4*)(Fr + 4);
    float4 f2 = *(const float4*)(Fr + 16);
    float4 f3 = *(const float4*)(Fr + 20);
    Fa[0]  = (_Float16)f0.x; Fa[1]  = (_Float16)f0.y;
    Fa[2]  = (_Float16)f0.z; Fa[3]  = (_Float16)f0.w;
    Fa[4]  = (_Float16)f1.x; Fa[5]  = (_Float16)f1.y;
    Fa[6]  = (_Float16)f1.z; Fa[7]  = (_Float16)f1.w;
    Fa[8]  = (_Float16)f2.x; Fa[9]  = (_Float16)f2.y;
    Fa[10] = (_Float16)f2.z; Fa[11] = (_Float16)f2.w;
    Fa[12] = (_Float16)f3.x; Fa[13] = (_Float16)f3.y;
    Fa[14] = (_Float16)f3.z; Fa[15] = (_Float16)f3.w;
#pragma unroll
    for (int nt = 0; nt < 4; ++nt) {
      const v8h* wp = (const v8h*)&WT[nt * 16 + n][kt * 32 + h * 8];
      v8h w0 = wp[0], w1 = wp[2];
      v16h Wb;
#pragma unroll
      for (int i = 0; i < 8; ++i) { Wb[i] = w0[i]; Wb[8 + i] = w1[i]; }
      Xc[nt] = __builtin_amdgcn_wmma_f32_16x16x32_f16(
          false, Fa, false, Wb, (short)0, Xc[nt], false, false);
    }
  }
  // bias + sigmoid, store transposed f16 tile into per-wave LDS
#pragma unroll
  for (int nt = 0; nt < 4; ++nt) {
    float bb = bin[nt * 16 + n];
    v8h p;
#pragma unroll
    for (int r = 0; r < 8; ++r) {
      float v = Xc[nt][r] + bb;
      p[r] = (_Float16)(1.0f / (1.0f + __expf(-v)));
    }
    *(v8h*)&XT[wave][nt * 16 + n][h * 8] = p;
  }

  // 3 random-walk steps: reduce xs, propagate x <- adj @ x (no nonlinearity)
#pragma unroll
  for (int step = 0; step < 3; ++step) {
#pragma unroll
    for (int half = 0; half < 2; ++half) {
      int c = lane + 32 * half;
      v8h x0 = *(const v8h*)&XT[wave][c][0];
      v8h x1 = *(const v8h*)&XT[wave][c][8];
      float s = 0.0f;
#pragma unroll
      for (int i = 0; i < 8; ++i) s += (float)x0[i] + (float)x1[i];
      XS[wave][step][c] = s;
    }
    if (step < 2) {
      v8f Xn[4];
#pragma unroll
      for (int nt = 0; nt < 4; ++nt) {
        v16h Xb;
#pragma unroll
        for (int i = 0; i < 16; ++i) Xb[i] = (_Float16)0.0f;
        {  // elems 0-7 hold K=8h..8h+7 (=s); elems 8-15 stay zero (pad K=16..31)
          v8h x0 = *(const v8h*)&XT[wave][nt * 16 + n][h * 8];
#pragma unroll
          for (int i = 0; i < 8; ++i) { Xb[i] = x0[i]; }
        }
        Xn[nt] = __builtin_amdgcn_wmma_f32_16x16x32_f16(
            false, adjA, false, Xb, (short)0, v8f_zero(), false, false);
      }
#pragma unroll
      for (int nt = 0; nt < 4; ++nt) {
        v8h p;
#pragma unroll
        for (int r = 0; r < 8; ++r) p[r] = (_Float16)Xn[nt][r];
        *(v8h*)&XT[wave][nt * 16 + n][h * 8] = p;
      }
    }
  }
  __syncthreads();

  // wave 0: out_tile(16x32) = sum_step xs_step(16x64) @ zs_step(64x32)
  if (wave == 0) {
    v8f Oc[2];
    Oc[0] = v8f_zero(); Oc[1] = v8f_zero();
#pragma unroll
    for (int step = 0; step < 3; ++step) {
#pragma unroll
      for (int kt = 0; kt < 2; ++kt) {
        v16h Af;
        const float4* xa = (const float4*)&XS[n][step][kt * 32 + h * 8];
        const float4* xb = (const float4*)&XS[n][step][kt * 32 + 16 + h * 8];
        float4 a0 = xa[0], a1 = xa[1], a2 = xb[0], a3 = xb[1];
        Af[0]  = (_Float16)a0.x; Af[1]  = (_Float16)a0.y;
        Af[2]  = (_Float16)a0.z; Af[3]  = (_Float16)a0.w;
        Af[4]  = (_Float16)a1.x; Af[5]  = (_Float16)a1.y;
        Af[6]  = (_Float16)a1.z; Af[7]  = (_Float16)a1.w;
        Af[8]  = (_Float16)a2.x; Af[9]  = (_Float16)a2.y;
        Af[10] = (_Float16)a2.z; Af[11] = (_Float16)a2.w;
        Af[12] = (_Float16)a3.x; Af[13] = (_Float16)a3.y;
        Af[14] = (_Float16)a3.z; Af[15] = (_Float16)a3.w;
#pragma unroll
        for (int ot = 0; ot < 2; ++ot) {
          const v8h* zp = (const v8h*)(zsT + ((size_t)(step * 32 + ot * 16 + n) * 64
                                              + kt * 32 + h * 8));
          v8h z0 = zp[0], z1 = zp[2];
          v16h Zb;
#pragma unroll
          for (int i = 0; i < 8; ++i) { Zb[i] = z0[i]; Zb[8 + i] = z1[i]; }
          Oc[ot] = __builtin_amdgcn_wmma_f32_16x16x32_f16(
              false, Af, false, Zb, (short)0, Oc[ot], false, false);
        }
      }
    }
    const float inv3 = 1.0f / 3.0f;
    const long b0 = (long)blockIdx.x * 16;
#pragma unroll
    for (int ot = 0; ot < 2; ++ot) {
      float s = 0.0f, s2 = 0.0f;
#pragma unroll
      for (int r = 0; r < 8; ++r) {
        float v = Oc[ot][r] * inv3;
        out_raw[(b0 + 8 * h + r) * 32 + ot * 16 + n] = v;
        s += v;
        s2 += v * v;
      }
      atomicAdd(&accum[ot * 16 + n], s);
      atomicAdd(&accum[32 + ot * 16 + n], s2);
    }
  }
}

// ---------------------------------------------------------------------------
// Kernel 3: BatchNorm (training-mode batch stats) + sigmoid, in place on d_out
// ---------------------------------------------------------------------------
__global__ __launch_bounds__(256) void k_bn(
    float* __restrict__ out, const float* __restrict__ accum,
    const float* __restrict__ gamma, const float* __restrict__ beta, int Bn)
{
  int i = blockIdx.x * blockDim.x + threadIdx.x;
  if (i >= Bn * 32) return;
  int ch = i & 31;
  float fB = (float)Bn;
  float mean = accum[ch] / fB;
  float var  = accum[32 + ch] / fB - mean * mean;
  float y = (out[i] - mean) * rsqrtf(var + 1e-5f) * gamma[ch] + beta[ch];
  out[i] = 1.0f / (1.0f + __expf(-y));
}

extern "C" void kernel_launch(void* const* d_in, const int* in_sizes, int n_in,
                              void* d_out, int out_size, void* d_ws, size_t ws_size,
                              hipStream_t stream) {
  const float* adj   = (const float*)d_in[0];
  const float* feat  = (const float*)d_in[1];
  const float* Win   = (const float*)d_in[2];
  const float* bin   = (const float*)d_in[3];
  const float* fh    = (const float*)d_in[4];
  const float* ah    = (const float*)d_in[5];
  const float* gamma = (const float*)d_in[6];
  const float* beta  = (const float*)d_in[7];
  float* out = (float*)d_out;

  const int Bn = in_sizes[0] / 256;               // adj is (B,16,16)

  _Float16* zsT = (_Float16*)d_ws;                // 3*32*64 f16 = 12 KB
  float* accum  = (float*)((char*)d_ws + 16384);  // 64 f32

  k_hidden<<<1, 256, 0, stream>>>(fh, ah, zsT, accum);
  k_rw_main<<<Bn / 16, 512, 0, stream>>>(adj, feat, Win, bin, zsT, accum, out);
  k_bn<<<(Bn * 32 + 255) / 256, 256, 0, stream>>>(out, accum, gamma, beta, Bn);
}
